// MPNNModel_A_T_17119739642177
// MI455X (gfx1250) — matrix operand order repaired and checked by reading, not compiled, not run on hardware
//
#include <hip/hip_runtime.h>
#include <hip/hip_bf16.h>

// ---------------------------------------------------------------------------
// MPNN on gfx1250: bf16 WMMA GEMMs + CSR gather (atomic-free edge phase).
// ---------------------------------------------------------------------------

#define DIM_H 128
#define N_LAYERS 5

typedef __attribute__((ext_vector_type(16))) __bf16 v16bf;
typedef __attribute__((ext_vector_type(8)))  float  v8f;

union ABFrag {
    uint4  q[2];
    v16bf  v;
};

__device__ __forceinline__ unsigned short f32_to_bf16_rne(float f) {
    unsigned int u = __float_as_uint(f);
    u += 0x7FFFu + ((u >> 16) & 1u);
    return (unsigned short)(u >> 16);
}

__device__ __forceinline__ float bf16_to_f32(unsigned short h) {
    return __uint_as_float(((unsigned int)h) << 16);
}

// ---------------------------------------------------------------------------
// Pack mpnn_w (20 x 128 x 128 f32) into per-(layer,type) B-fragment layout.
// Dest idx = (((ij*8 + nt)*4 + kk)*32 + lane)*16 + e
// B elem:   K = kk*32 + e + 8*(lane>>4) + (e>=8 ? 8:0) ; N = nt*16 + (lane&15)
// ---------------------------------------------------------------------------
__global__ void pack_w_kernel(const float* __restrict__ mw,
                              unsigned short* __restrict__ wp, int total) {
    int idx = blockIdx.x * 256 + threadIdx.x;
    if (idx >= total) return;
    int e    = idx & 15;
    int lane = (idx >> 4) & 31;
    int kk   = (idx >> 9) & 3;
    int nt   = (idx >> 11) & 7;
    int ij   = idx >> 14;
    int half = lane >> 4;
    int K = kk * 32 + e + 8 * half + (e >= 8 ? 8 : 0);
    int N = nt * 16 + (lane & 15);
    wp[idx] = f32_to_bf16_rne(mw[((size_t)ij * 128 + K) * 128 + N]);
}

// ---------------------------------------------------------------------------
// Pack x (Mrows x 128 f32) into A-fragment layout (optionally fused ReLU).
// Dest idx = (((rt*4 + kk)*32 + lane)*16 + e
// A elem:   M = rt*16 + (lane&15) ; K = kk*32 + e + 8*(lane>>4) + (e>=8 ? 8:0)
// ---------------------------------------------------------------------------
__global__ void pack_x_kernel(const float* __restrict__ x,
                              unsigned short* __restrict__ xp,
                              int mtiles, int Mrows, int do_relu) {
    size_t idx = (size_t)blockIdx.x * 256 + threadIdx.x;
    if (idx >= (size_t)mtiles * 2048) return;
    int e    = (int)(idx & 15);
    int lane = (int)((idx >> 4) & 31);
    int kk   = (int)((idx >> 9) & 3);
    int rt   = (int)(idx >> 11);
    int half = lane >> 4;
    int M = rt * 16 + (lane & 15);
    int K = kk * 32 + e + 8 * half + (e >= 8 ? 8 : 0);
    float v = 0.0f;
    if (M < Mrows) {
        v = x[(size_t)M * 128 + K];
        if (do_relu) v = fmaxf(v, 0.0f);
    }
    xp[idx] = f32_to_bf16_rne(v);
}

// ---------------------------------------------------------------------------
// m = relu(X @ W + b) via v_wmma_f32_16x16x32_bf16, stored bf16 row-major.
// grid.x = row tiles; block = 256 (8 waves); wave w -> column tile w.
// ---------------------------------------------------------------------------
__global__ void gemm_relu_kernel(const unsigned short* __restrict__ xp,
                                 const unsigned short* __restrict__ wp,
                                 const float* __restrict__ bias,
                                 unsigned short* __restrict__ m,
                                 int Mrows) {
    int rt   = blockIdx.x;
    int wave = threadIdx.x >> 5;   // N-tile 0..7
    int lane = threadIdx.x & 31;

    v8f c = {};
#pragma unroll
    for (int kk = 0; kk < 4; ++kk) {
        ABFrag a, b;
        const uint4* ap =
            (const uint4*)(xp + (((size_t)rt * 4 + kk) * 32 + lane) * 16);
        a.q[0] = ap[0];
        a.q[1] = ap[1];
        const uint4* bp =
            (const uint4*)(wp + ((size_t)wave * 4 + kk) * 512 + (size_t)lane * 16);
        b.q[0] = bp[0];
        b.q[1] = bp[1];
        c = __builtin_amdgcn_wmma_f32_16x16x32_bf16(
                false, a.v, false, b.v, (short)0, c, false, false);
    }

    int col  = wave * 16 + (lane & 15);
    int half = lane >> 4;
    float bb = bias[col];
    int rowBase = rt * 16 + 8 * half;
    unsigned short* mrow = m + (size_t)rowBase * 128 + col;

    if (rt * 16 + 16 <= Mrows) {
        // full tile: unguarded clause of 8 stores
#pragma unroll
        for (int r = 0; r < 8; ++r) {
            float v = fmaxf(c[r] + bb, 0.0f);
            mrow[(size_t)r * 128] = f32_to_bf16_rne(v);
        }
    } else {
#pragma unroll
        for (int r = 0; r < 8; ++r) {
            if (rowBase + r < Mrows) {
                float v = fmaxf(c[r] + bb, 0.0f);
                mrow[(size_t)r * 128] = f32_to_bf16_rne(v);
            }
        }
    }
}

// ---------------------------------------------------------------------------
// CSR build: count -> two-level exclusive scan -> stable-ish fill.
// ---------------------------------------------------------------------------
__global__ void count_kernel(const int* __restrict__ tgt, int* __restrict__ counts,
                             int E) {
    int e = blockIdx.x * 256 + threadIdx.x;
    if (e < E) atomicAdd(&counts[tgt[e]], 1);
}

__global__ void reduce256_kernel(const int* __restrict__ v, int* __restrict__ bsum,
                                 int N) {
    __shared__ int s[256];
    int i = blockIdx.x * 256 + threadIdx.x;
    s[threadIdx.x] = (i < N) ? v[i] : 0;
    __syncthreads();
    for (int d = 128; d > 0; d >>= 1) {
        if (threadIdx.x < d) s[threadIdx.x] += s[threadIdx.x + d];
        __syncthreads();
    }
    if (threadIdx.x == 0) bsum[blockIdx.x] = s[0];
}

__global__ void scan_sums_kernel(int* __restrict__ bsum, int nb) {
    __shared__ int s[512];
    int tid = threadIdx.x;
    int v = (tid < nb) ? bsum[tid] : 0;
    s[tid] = v;
    __syncthreads();
    for (int d = 1; d < 512; d <<= 1) {
        int t = (tid >= d) ? s[tid - d] : 0;
        __syncthreads();
        s[tid] += t;
        __syncthreads();
    }
    if (tid < nb) bsum[tid] = s[tid] - v;  // exclusive
}

__global__ void scan_blocks_kernel(const int* __restrict__ v,
                                   const int* __restrict__ bexcl,
                                   int* __restrict__ off, int N) {
    __shared__ int s[256];
    int tid = threadIdx.x;
    int i = blockIdx.x * 256 + tid;
    int val = (i < N) ? v[i] : 0;
    s[tid] = val;
    __syncthreads();
    for (int d = 1; d < 256; d <<= 1) {
        int t = (tid >= d) ? s[tid - d] : 0;
        __syncthreads();
        s[tid] += t;
        __syncthreads();
    }
    int incl = s[tid];
    int base = bexcl[blockIdx.x];
    if (i < N) {
        off[i] = base + incl - val;
        if (i == N - 1) off[N] = base + incl;
    }
}

__global__ void fill_kernel(const int* __restrict__ tgt,
                            const int* __restrict__ off,
                            int* __restrict__ cursor,
                            int* __restrict__ perm, int E) {
    int e = blockIdx.x * 256 + threadIdx.x;
    if (e < E) {
        int tg = tgt[e];
        int p = atomicAdd(&cursor[tg], 1);
        perm[off[tg] + p] = e;
    }
}

// ---------------------------------------------------------------------------
// Edge gather: out[n] (+)= sum_{e in in-edges(n)} m[src(e)] * w(e)
// One wave per target node; 32 lanes x 4 features; accumulate in VGPRs.
// ---------------------------------------------------------------------------
__global__ void edge_gather_kernel(const int* __restrict__ off,
                                   const int* __restrict__ perm,
                                   const int* __restrict__ es,  // src ids [0,E)
                                   const float* __restrict__ ew,
                                   const unsigned short* __restrict__ m,
                                   float* __restrict__ out,
                                   int Ntgt, int accumulate) {
    int node = blockIdx.x * 8 + (threadIdx.x >> 5);
    if (node >= Ntgt) return;
    int lane = threadIdx.x & 31;

    float a0 = 0.0f, a1 = 0.0f, a2 = 0.0f, a3 = 0.0f;
    int beg = off[node];
    int end = off[node + 1];
    for (int k = beg; k < end; ++k) {
        int e = perm[k];
        int src = es[e];
        float w = ew[e];
        const uint2 raw = *(const uint2*)(m + (size_t)src * 128 + lane * 4);
        a0 += bf16_to_f32((unsigned short)(raw.x & 0xFFFFu)) * w;
        a1 += bf16_to_f32((unsigned short)(raw.x >> 16)) * w;
        a2 += bf16_to_f32((unsigned short)(raw.y & 0xFFFFu)) * w;
        a3 += bf16_to_f32((unsigned short)(raw.y >> 16)) * w;
    }

    float4* orow = (float4*)(out + (size_t)node * 128 + lane * 4);
    if (accumulate) {
        float4 prev = *orow;
        prev.x += a0; prev.y += a1; prev.z += a2; prev.w += a3;
        *orow = prev;
    } else {
        float4 res = {a0, a1, a2, a3};
        *orow = res;
    }
}

// ---------------------------------------------------------------------------
// Encoder: out0[n][h] = enc_al_w[al[n]][h] + sum_k ac[n][k]*enc_ac_w[k][h] + b
// ---------------------------------------------------------------------------
__global__ void encoder_kernel(const int* __restrict__ al,
                               const float* __restrict__ ac,
                               const float* __restrict__ enc_al_w,
                               const float* __restrict__ enc_ac_w,
                               const float* __restrict__ enc_ac_b,
                               float* __restrict__ out0, int N0) {
    __shared__ float sac[64];
    int n = blockIdx.x;
    if (n >= N0) return;
    int h = threadIdx.x;
    if (h < 64) sac[h] = ac[(size_t)n * 64 + h];
    __syncthreads();
    float acc = enc_ac_b[h] + enc_al_w[(size_t)al[n] * 128 + h];
#pragma unroll 8
    for (int k = 0; k < 64; ++k) acc += sac[k] * enc_ac_w[(size_t)k * 128 + h];
    out0[(size_t)n * 128 + h] = acc;
}

__global__ void init_x1_kernel(const int* __restrict__ t,
                               const float* __restrict__ emb,
                               float* __restrict__ out1, int N1) {
    int n = blockIdx.x;
    if (n >= N1) return;
    int h = threadIdx.x;
    out1[(size_t)n * 128 + h] = emb[(size_t)t[n] * 128 + h];
}

// ---------------------------------------------------------------------------
// Decoder: last = relu(out0) @ dec_w + dec_b ; softmax over 3 classes.
// ---------------------------------------------------------------------------
__global__ void decoder_kernel(const float* __restrict__ out0,
                               const float* __restrict__ dec_w,  // 128x3
                               const float* __restrict__ dec_b,  // 3
                               float* __restrict__ dout, int N0) {
    int node = blockIdx.x * 8 + (threadIdx.x >> 5);
    if (node >= N0) return;
    int lane = threadIdx.x & 31;
    const float4 xv = *(const float4*)(out0 + (size_t)node * 128 + lane * 4);
    float x[4] = {fmaxf(xv.x, 0.0f), fmaxf(xv.y, 0.0f),
                  fmaxf(xv.z, 0.0f), fmaxf(xv.w, 0.0f)};
    float acc[3];
#pragma unroll
    for (int cc = 0; cc < 3; ++cc) {
        float s = 0.0f;
#pragma unroll
        for (int kk = 0; kk < 4; ++kk)
            s += x[kk] * dec_w[(size_t)(lane * 4 + kk) * 3 + cc];
        acc[cc] = s;
    }
#pragma unroll
    for (int off = 16; off >= 1; off >>= 1) {
#pragma unroll
        for (int cc = 0; cc < 3; ++cc)
            acc[cc] += __shfl_xor(acc[cc], off, 32);
    }
    if (lane == 0) {
        float l0 = acc[0] + dec_b[0];
        float l1 = acc[1] + dec_b[1];
        float l2 = acc[2] + dec_b[2];
        dout[(size_t)node * 3 + 0] = l0;
        dout[(size_t)node * 3 + 1] = l1;
        dout[(size_t)node * 3 + 2] = l2;
        float mx = fmaxf(l0, fmaxf(l1, l2));
        float e0 = __expf(l0 - mx), e1 = __expf(l1 - mx), e2 = __expf(l2 - mx);
        float inv = 1.0f / (e0 + e1 + e2);
        float* sm = dout + (size_t)N0 * 3;
        sm[(size_t)node * 3 + 0] = e0 * inv;
        sm[(size_t)node * 3 + 1] = e1 * inv;
        sm[(size_t)node * 3 + 2] = e2 * inv;
    }
}

// ---------------------------------------------------------------------------

static inline size_t align256(size_t x) { return (x + 255) & ~(size_t)255; }

extern "C" void kernel_launch(void* const* d_in, const int* in_sizes, int n_in,
                              void* d_out, int out_size, void* d_ws, size_t ws_size,
                              hipStream_t stream) {
    const int*   al        = (const int*)d_in[0];
    const float* ac        = (const float*)d_in[1];
    const int*   t         = (const int*)d_in[2];
    const int*   es[4]     = {(const int*)d_in[3], (const int*)d_in[5],
                              (const int*)d_in[7], (const int*)d_in[9]};
    const float* ew[4]     = {(const float*)d_in[4], (const float*)d_in[6],
                              (const float*)d_in[8], (const float*)d_in[10]};
    const float* enc_al_w  = (const float*)d_in[11];
    const float* enc_ac_w  = (const float*)d_in[12];
    const float* enc_ac_b  = (const float*)d_in[13];
    const float* emb_test  = (const float*)d_in[14];
    const float* mpnn_w    = (const float*)d_in[15];
    const float* mpnn_b    = (const float*)d_in[16];
    const float* dec_w     = (const float*)d_in[17];
    const float* dec_b     = (const float*)d_in[18];

    const int N0 = in_sizes[0];
    const int N1 = in_sizes[2];
    const int Ej[4] = {in_sizes[4], in_sizes[6], in_sizes[8], in_sizes[10]};
    const int T_SRCS[4] = {0, 1, 0, 1};
    const int T_TGTS[4] = {0, 0, 1, 1};

    const int mt0 = (N0 + 15) / 16;
    const int mt1 = (N1 + 15) / 16;
    const int Nmax = (N0 > N1) ? N0 : N1;

    // workspace carve-out
    char* ws = (char*)d_ws;
    size_t off = 0;
    float* out0 = (float*)(ws + off);          off += align256((size_t)N0 * 128 * 4);
    float* out1 = (float*)(ws + off);          off += align256((size_t)N1 * 128 * 4);
    unsigned short* x0p = (unsigned short*)(ws + off);  off += align256((size_t)mt0 * 2048 * 2);
    unsigned short* x1p = (unsigned short*)(ws + off);  off += align256((size_t)mt1 * 2048 * 2);
    unsigned short* mbuf = (unsigned short*)(ws + off); off += align256((size_t)mt0 * 16 * 128 * 2);
    unsigned short* wp   = (unsigned short*)(ws + off); off += align256((size_t)20 * 16384 * 2);
    int* csr_off[4];
    int* csr_perm[4];
    for (int j = 0; j < 4; ++j) {
        const int Nt = (T_TGTS[j] == 0) ? N0 : N1;
        csr_off[j]  = (int*)(ws + off); off += align256((size_t)(Nt + 1) * 4);
        csr_perm[j] = (int*)(ws + off); off += align256((size_t)Ej[j] * 4);
    }
    int* counts = (int*)(ws + off); off += align256((size_t)Nmax * 4);
    int* bsum   = (int*)(ws + off); off += align256((size_t)512 * 4);

    // 1) pack all layer weights to bf16 B-fragment layout
    {
        int total = 20 * 16384;
        pack_w_kernel<<<(total + 255) / 256, 256, 0, stream>>>(mpnn_w, wp, total);
    }

    // 2) build CSR-by-target for each edge type (graph is layer-invariant)
    for (int j = 0; j < 4; ++j) {
        const int Nt = (T_TGTS[j] == 0) ? N0 : N1;
        const int E  = Ej[j];
        const int* tgt = es[j] + E;  // es stacked [src | tgt]
        const int nbE = (E + 255) / 256;
        const int nbN = (Nt + 255) / 256;
        hipMemsetAsync(counts, 0, (size_t)Nt * 4, stream);
        count_kernel<<<nbE, 256, 0, stream>>>(tgt, counts, E);
        reduce256_kernel<<<nbN, 256, 0, stream>>>(counts, bsum, Nt);
        scan_sums_kernel<<<1, 512, 0, stream>>>(bsum, nbN);
        scan_blocks_kernel<<<nbN, 256, 0, stream>>>(counts, bsum, csr_off[j], Nt);
        hipMemsetAsync(counts, 0, (size_t)Nt * 4, stream);
        fill_kernel<<<nbE, 256, 0, stream>>>(tgt, csr_off[j], counts,
                                             csr_perm[j], E);
    }

    // 3) encode node features into out0/out1 (acting as x-buffers for layer 0)
    encoder_kernel<<<N0, 128, 0, stream>>>(al, ac, enc_al_w, enc_ac_w, enc_ac_b,
                                           out0, N0);
    init_x1_kernel<<<N1, 128, 0, stream>>>(t, emb_test, out1, N1);

    // 4) message-passing layers
    for (int i = 0; i < N_LAYERS; ++i) {
        const int do_relu = (i > 0) ? 1 : 0;
        {
            size_t tot0 = (size_t)mt0 * 2048;
            pack_x_kernel<<<(unsigned)((tot0 + 255) / 256), 256, 0, stream>>>(
                out0, x0p, mt0, N0, do_relu);
            size_t tot1 = (size_t)mt1 * 2048;
            pack_x_kernel<<<(unsigned)((tot1 + 255) / 256), 256, 0, stream>>>(
                out1, x1p, mt1, N1, do_relu);
        }

        for (int j = 0; j < 4; ++j) {
            const int ij = i * 4 + j;
            const unsigned short* xp = (T_SRCS[j] == 0) ? x0p : x1p;
            const int srcRows  = (T_SRCS[j] == 0) ? N0 : N1;
            const int srcTiles = (T_SRCS[j] == 0) ? mt0 : mt1;
            float* outT = (T_TGTS[j] == 0) ? out0 : out1;
            const int Nt = (T_TGTS[j] == 0) ? N0 : N1;
            const int accum = (j == 1 || j == 3) ? 1 : 0;  // second type per tgt

            gemm_relu_kernel<<<srcTiles, 256, 0, stream>>>(
                xp, wp + (size_t)ij * 16384, mpnn_b + (size_t)ij * 128,
                mbuf, srcRows);

            edge_gather_kernel<<<(Nt + 7) / 8, 256, 0, stream>>>(
                csr_off[j], csr_perm[j], es[j], ew[j], mbuf, outT, Nt, accum);
        }
    }

    // 5) decode + softmax (ReLU fused on read of out0)
    decoder_kernel<<<(N0 + 7) / 8, 256, 0, stream>>>(out0, dec_w, dec_b,
                                                     (float*)d_out, N0);
}
